// LocalMQA_88021059764274
// MI455X (gfx1250) — matrix operand order repaired and checked by reading, not compiled, not run on hardware
//
#include <hip/hip_runtime.h>
#include <hip/hip_bf16.h>

#define SEQ_LEN 4096
#define D_MODEL 768
#define N_HEADS 8
#define QD 96
#define WINDOW 128
#define QKV_DIM 960  // QD*(N_HEADS+2)

typedef __attribute__((ext_vector_type(16))) _Float16 v16h;
typedef __attribute__((ext_vector_type(8)))  _Float16 v8h;
typedef __attribute__((ext_vector_type(8)))  float    v8f;

#define WMMA_F16(a, b, c) \
  __builtin_amdgcn_wmma_f32_16x16x32_f16(false, (a), false, (b), (short)0, (c), false, false)

__device__ __forceinline__ int lane_id() { return (int)(threadIdx.x & 31u); }

// Async global->LDS copy of 16B per lane (GLOBAL_LOAD_ASYNC_TO_LDS_B128,
// tracked by ASYNCcnt).  lds_off = wave-relative LDS byte offset (low 32 bits
// of a generic pointer to __shared__, per the CDNA5 LDS aperture rule).
__device__ __forceinline__ void async_g2l_b128(unsigned int lds_off, const void* gptr) {
  asm volatile("global_load_async_to_lds_b128 %0, %1, off"
               :: "v"(lds_off), "v"((unsigned long long)(uintptr_t)gptr)
               : "memory");
}
__device__ __forceinline__ void wait_async0() {
  asm volatile("s_wait_asynccnt 0x0" ::: "memory");
}

// ---------------------------------------------------------------------------
// Fragment loaders (wave32, V_WMMA_F32_16X16X32_F16; A/B use this code's contiguous k order)
// A (16x32): lane l&15 = row m; elems 0..15 <- K=16*(l>>4)+e (contiguous within lane)
// B (32x16) with B[k][n] = W[n][k]: lane l&15 = n (a W row);
//           elems 0..15 <- K = 16*(l>>4)+e (contiguous within lane)
// C/D (16x16 f32): lane l&15 = n; elem r -> m = r + 8*(l>>4)
// ---------------------------------------------------------------------------
__device__ __forceinline__ v16h load_afrag(const _Float16* __restrict__ base, int ld) {
  const int l = lane_id();
  const _Float16* p = base + (l & 15) * ld + ((l >> 4) << 4);
  union { v16h v; v8h h[2]; } u;
  u.h[0] = *(const v8h*)(p);
  u.h[1] = *(const v8h*)(p + 8);
  return u.v;
}

__device__ __forceinline__ v16h load_bfrag(const _Float16* base, int ld) {
  const int l = lane_id();
  const _Float16* p = base + (l & 15) * ld + ((l >> 4) << 4);
  union { v16h v; v8h h[2]; } u;
  u.h[0] = *(const v8h*)(p);
  u.h[1] = *(const v8h*)(p + 8);
  return u.v;
}

// B-fragment with per-lane row clamp (rows index key positions, may be OOB; the
// score mask kills those columns, the clamp just keeps the address legal).
__device__ __forceinline__ v16h load_bfrag_rowclamp(const _Float16* __restrict__ base,
                                                    int ld, int row0, int maxrow) {
  const int l = lane_id();
  int row = row0 + (l & 15);
  row = row < 0 ? 0 : (row > maxrow ? maxrow : row);
  const _Float16* p = base + row * ld + ((l >> 4) << 4);
  union { v16h v; v8h h[2]; } u;
  u.h[0] = *(const v8h*)(p);
  u.h[1] = *(const v8h*)(p + 8);
  return u.v;
}

// B-fragment with clamped (8-aligned) column chunk starts; out-of-window
// columns correspond to zero P entries so clamped garbage contributes 0.
__device__ __forceinline__ v16h load_bfrag_colclamp(const _Float16* __restrict__ base,
                                                    int ld, int col0, int maxstart) {
  const int l = lane_id();
  const int row = l & 15;
  int c0 = col0 + ((l >> 4) << 4);
  int c1 = c0 + 8;
  c0 = c0 < 0 ? 0 : (c0 > maxstart ? maxstart : c0);
  c1 = c1 < 0 ? 0 : (c1 > maxstart ? maxstart : c1);
  union { v16h v; v8h h[2]; } u;
  u.h[0] = *(const v8h*)(base + row * ld + c0);
  u.h[1] = *(const v8h*)(base + row * ld + c1);
  return u.v;
}

// ---------------------------------------------------------------------------
// fp32 -> fp16 conversion
// ---------------------------------------------------------------------------
__global__ void __launch_bounds__(256) cvt_f32_f16(const float* __restrict__ in,
                                                   _Float16* __restrict__ out, int n) {
  int i = blockIdx.x * 256 + threadIdx.x;
  if (i < n) out[i] = (_Float16)in[i];
}

// ---------------------------------------------------------------------------
// C[M][N] = A[M][K](f16) * W[N][K](f16)^T + bias[N]   (fp32 out), K % 64 == 0.
// block = 128 threads = 4 waves (2x2), wave tile 32x32, block tile 64x64.
// The 64x32 W tile of each K-step is staged into LDS with
// GLOBAL_LOAD_ASYNC_TO_LDS_B128, double buffered (DMA for step k+1 overlaps
// the WMMAs of step k).  K-loop is unrolled 2x so each half references a
// compile-time buffer => immediate ds_load offsets.  Per-thread DMA addresses
// are precomputed; each issue is one pointer add + one async instruction.
// ---------------------------------------------------------------------------
__global__ void __launch_bounds__(128) gemm_xwt(const _Float16* __restrict__ A,
                                                const _Float16* __restrict__ W,
                                                const float* __restrict__ bias,
                                                float* __restrict__ C,
                                                int M, int N, int K) {
  __shared__ _Float16 Wt[2][64][32];  // 8 KB double buffer (4096 B per buffer)

  const int t    = (int)threadIdx.x;
  const int wave = t >> 5;
  const int lane = lane_id();
  const int m0   = blockIdx.x * 64 + (wave >> 1) * 32;
  const int nblk = blockIdx.y * 64;
  const int wn   = (wave & 1) * 32;

  v8f c00 = {}, c01 = {}, c10 = {}, c11 = {};
  const _Float16* Ab0 = A + (size_t)m0 * K;
  const _Float16* Ab1 = A + (size_t)(m0 + 16) * K;

  // Per-thread DMA slots: thread t moves W rows (t>>2) and (t>>2)+32 of the
  // 64-row tile, 16B chunk (t&3).  256 chunks of 16B = 4 KB per K-step.
  const int drow = t >> 2;
  const int dq   = (t & 3) << 3;  // halves: 0,8,16,24
  const _Float16* g0 = W + (size_t)(nblk + drow) * K + dq;
  const _Float16* g1 = W + (size_t)(nblk + drow + 32) * K + dq;
  const unsigned  l0 = (unsigned)(uintptr_t)(&Wt[0][drow][dq]);
  const unsigned  l1 = (unsigned)(uintptr_t)(&Wt[0][drow + 32][dq]);

  auto issueW = [&](int k, unsigned bufByte) {
    async_g2l_b128(l0 + bufByte, g0 + k);
    async_g2l_b128(l1 + bufByte, g1 + k);
  };

  auto compute = [&](int k, const _Float16* wb) {
    __builtin_prefetch((const void*)(Ab0 + k + 256), 0, 1);  // global_prefetch_b8
    v16h a0 = load_afrag(Ab0 + k, K);
    v16h a1 = load_afrag(Ab1 + k, K);
    v16h b0 = load_bfrag(wb, 32);            // ds_load_b128, immediate offsets
    v16h b1 = load_bfrag(wb + 16 * 32, 32);
    c00 = WMMA_F16(a0, b0, c00);
    c01 = WMMA_F16(a0, b1, c01);
    c10 = WMMA_F16(a1, b0, c10);
    c11 = WMMA_F16(a1, b1, c11);
  };

  issueW(0, 0);
  for (int k = 0; k < K; k += 64) {
    wait_async0();        // this wave's DMAs into buf0 landed in LDS
    __syncthreads();      // everyone's landed; buf1 reads of prev step retired
    if (k + 32 < K) issueW(k + 32, 4096u);
    compute(k, &Wt[0][wn][0]);

    wait_async0();
    __syncthreads();
    if (k + 64 < K) issueW(k + 64, 0u);
    compute(k + 32, &Wt[1][wn][0]);
  }

  const int n0 = nblk + wn;
  const int nl = lane & 15, hf = lane >> 4;
  const float bi0 = bias[n0 + nl];
  const float bi1 = bias[n0 + 16 + nl];
#pragma unroll
  for (int r = 0; r < 8; ++r) {
    int m = r + 8 * hf;
    C[(size_t)(m0 + m) * N + n0 + nl]           = c00[r] + bi0;
    C[(size_t)(m0 + m) * N + n0 + 16 + nl]      = c01[r] + bi1;
    C[(size_t)(m0 + 16 + m) * N + n0 + nl]      = c10[r] + bi0;
    C[(size_t)(m0 + 16 + m) * N + n0 + 16 + nl] = c11[r] + bi1;
  }
}

// ---------------------------------------------------------------------------
// RoPE + split qkv(f32)[L][960] -> kh(f16)[L][96], vt(f16)[96][L], qh(f16)[8][L][96]
// ---------------------------------------------------------------------------
__global__ void __launch_bounds__(128) rope_split(const float* __restrict__ qkv,
                                                  _Float16* __restrict__ kh,
                                                  _Float16* __restrict__ vt,
                                                  _Float16* __restrict__ qh) {
  const int i = blockIdx.x;
  const int tid = (int)threadIdx.x;
  const float* row = qkv + (size_t)i * QKV_DIM;
  const float fi = (float)i;
  const float LN1E4 = 9.210340371976184f;  // ln(10000)

  for (int p = tid; p < 48; p += 128) {
    float th = __expf(-(2.0f * (float)p / 96.0f) * LN1E4);
    float sn, cs;
    __sincosf(fi * th, &sn, &cs);
    float e = row[2 * p], o = row[2 * p + 1];
    kh[(size_t)i * QD + 2 * p]     = (_Float16)(e * cs - o * sn);
    kh[(size_t)i * QD + 2 * p + 1] = (_Float16)(e * sn + o * cs);
  }
  for (int d = tid; d < QD; d += 128) {
    vt[(size_t)d * SEQ_LEN + i] = (_Float16)row[QD + d];
  }
  for (int idx = tid; idx < N_HEADS * 48; idx += 128) {
    int h = idx / 48, p = idx - h * 48;
    float th = __expf(-(2.0f * (float)p / 96.0f) * LN1E4);
    float sn, cs;
    __sincosf(fi * th, &sn, &cs);
    float e = row[2 * QD + h * QD + 2 * p];
    float o = row[2 * QD + h * QD + 2 * p + 1];
    size_t ob = ((size_t)h * SEQ_LEN + i) * QD + 2 * p;
    qh[ob]     = (_Float16)(e * cs - o * sn);
    qh[ob + 1] = (_Float16)(e * sn + o * cs);
  }
}

// ---------------------------------------------------------------------------
// Windowed MQA attention. One wave per (head, 16-query tile).
// 9 key tiles of 16 cover window j in (i-128, i]. Scores via WMMA, softmax in
// LDS (fp32), P kept f16 padded to 160 cols (5 K-steps of 32), PV via WMMA.
// ---------------------------------------------------------------------------
__global__ void __launch_bounds__(128) attn_kernel(const _Float16* __restrict__ qh,
                                                   const _Float16* __restrict__ kh,
                                                   const _Float16* __restrict__ vt,
                                                   _Float16* __restrict__ ao) {
  __shared__ float    S[4][16][160];
  __shared__ _Float16 P[4][16][160];
  __shared__ float    invs[4][16];

  const int w = (int)(threadIdx.x >> 5);
  const int lane = lane_id();
  const int task = blockIdx.x * 4 + w;       // 2048 tasks = 8 heads * 256 tiles
  const int h  = task >> 8;
  const int i0 = (task & 255) * 16;
  const int nl = lane & 15, hf = lane >> 4;

  const _Float16* qb = qh + ((size_t)h * SEQ_LEN + i0) * QD;
  v16h aq0 = load_afrag(qb, QD);
  v16h aq1 = load_afrag(qb + 32, QD);
  v16h aq2 = load_afrag(qb + 64, QD);

  // ---- scores ----
  for (int kt = 0; kt < 9; ++kt) {
    const int j0 = i0 - WINDOW + kt * 16;
    v16h bk0 = load_bfrag_rowclamp(kh,      QD, j0, SEQ_LEN - 1);
    v16h bk1 = load_bfrag_rowclamp(kh + 32, QD, j0, SEQ_LEN - 1);
    v16h bk2 = load_bfrag_rowclamp(kh + 64, QD, j0, SEQ_LEN - 1);
    v8f c = {};
    c = WMMA_F16(aq0, bk0, c);
    c = WMMA_F16(aq1, bk1, c);
    c = WMMA_F16(aq2, bk2, c);
    const int j = j0 + nl;
#pragma unroll
    for (int r = 0; r < 8; ++r) {
      int m = i0 + r + 8 * hf;
      bool ok = (j >= 0) && (j <= m) && (j > m - WINDOW);
      S[w][r + 8 * hf][kt * 16 + nl] = ok ? c[r] * 9.797958971132712f /*sqrt(96)*/
                                          : -1.0e30f;
    }
  }

  // ---- softmax (lanes 0..15, one row each; same-wave LDS, no barrier needed)
  if (lane < 16) {
    const int m = lane;
    float mx = -1.0e30f;
    for (int cidx = 0; cidx < 144; ++cidx) mx = fmaxf(mx, S[w][m][cidx]);
    float sum = 0.0f;
    for (int cidx = 0; cidx < 144; ++cidx) {
      float e = __expf(S[w][m][cidx] - mx);
      sum += e;
      P[w][m][cidx] = (_Float16)e;
    }
    for (int cidx = 144; cidx < 160; ++cidx) P[w][m][cidx] = (_Float16)0.0f;
    invs[w][m] = 1.0f / sum;
  }

  // ---- PV: out(16x96) = P(16x160) * V(160x96); V columns from vt[96][L] ----
  const int jlow = i0 - WINDOW;
  const _Float16* pb = &P[w][0][0];
  for (int nt = 0; nt < 6; ++nt) {
    const int d0 = nt * 16;
    v8f c = {};
#pragma unroll
    for (int s = 0; s < 5; ++s) {
      union { v16h v; v8h h8[2]; } u;   // A fragment of P from LDS (ds_load)
      const _Float16* pp = pb + (lane & 15) * 160 + 32 * s + ((lane >> 4) << 4);
      u.h8[0] = *(const v8h*)(pp);
      u.h8[1] = *(const v8h*)(pp + 8);
      v16h bv = load_bfrag_colclamp(vt + (size_t)d0 * SEQ_LEN, SEQ_LEN,
                                    jlow + 32 * s, SEQ_LEN - 8);
      c = WMMA_F16(u.v, bv, c);
    }
#pragma unroll
    for (int r = 0; r < 8; ++r) {
      int m = r + 8 * hf;
      float o = c[r] * invs[w][m];
      ao[(size_t)(i0 + m) * D_MODEL + h * QD + d0 + nl] = (_Float16)o;
    }
  }
}

// ---------------------------------------------------------------------------
// Launcher
// ---------------------------------------------------------------------------
extern "C" void kernel_launch(void* const* d_in, const int* in_sizes, int n_in,
                              void* d_out, int out_size, void* d_ws, size_t ws_size,
                              hipStream_t stream) {
  const float* x    = (const float*)d_in[0];  // [4096][768]
  const float* Wqkv = (const float*)d_in[1];  // [960][768]
  const float* bqkv = (const float*)d_in[2];  // [960]
  const float* Wout = (const float*)d_in[3];  // [768][768]
  const float* bout = (const float*)d_in[4];  // [768]
  float* out = (float*)d_out;                 // [4096][768]

  char* ws = (char*)d_ws;
  size_t off = 0;
  auto alloc = [&](size_t bytes) -> char* {
    char* p = ws + off;
    off += (bytes + 255) & ~(size_t)255;
    return p;
  };

  _Float16* xh    = (_Float16*)alloc((size_t)SEQ_LEN * D_MODEL * 2);
  _Float16* wqkvh = (_Float16*)alloc((size_t)QKV_DIM * D_MODEL * 2);
  _Float16* wouth = (_Float16*)alloc((size_t)D_MODEL * D_MODEL * 2);
  float*    qkv   = (float*)   alloc((size_t)SEQ_LEN * QKV_DIM * 4);
  _Float16* kh    = (_Float16*)alloc((size_t)SEQ_LEN * QD * 2);
  _Float16* vt    = (_Float16*)alloc((size_t)QD * SEQ_LEN * 2);
  _Float16* qhb   = (_Float16*)alloc((size_t)N_HEADS * SEQ_LEN * QD * 2);
  _Float16* ao    = (_Float16*)alloc((size_t)SEQ_LEN * D_MODEL * 2);

  const int nx  = SEQ_LEN * D_MODEL;
  const int nw1 = QKV_DIM * D_MODEL;
  const int nw2 = D_MODEL * D_MODEL;
  cvt_f32_f16<<<(nx + 255) / 256, 256, 0, stream>>>(x, xh, nx);
  cvt_f32_f16<<<(nw1 + 255) / 256, 256, 0, stream>>>(Wqkv, wqkvh, nw1);
  cvt_f32_f16<<<(nw2 + 255) / 256, 256, 0, stream>>>(Wout, wouth, nw2);

  // qkv = x @ Wqkv^T + bqkv
  gemm_xwt<<<dim3(SEQ_LEN / 64, QKV_DIM / 64), 128, 0, stream>>>(
      xh, wqkvh, bqkv, qkv, SEQ_LEN, QKV_DIM, D_MODEL);

  rope_split<<<SEQ_LEN, 128, 0, stream>>>(qkv, kh, vt, qhb);

  // 2048 wave-tasks, 4 waves per block
  attn_kernel<<<512, 128, 0, stream>>>(qhb, kh, vt, ao);

  // out = ao @ Wout^T + bout
  gemm_xwt<<<dim3(SEQ_LEN / 64, D_MODEL / 64), 128, 0, stream>>>(
      ao, wouth, bout, out, SEQ_LEN, D_MODEL, D_MODEL);
}
